// GRU_35957466202560
// MI455X (gfx1250) — matrix operand order repaired and checked by reading, not compiled, not run on hardware
//
#include <hip/hip_runtime.h>

#define EMB 1024
#define HID 1024
#define VOCAB 32000
#define SEQ 64
#define BATCH 64
#define L 2

typedef __attribute__((ext_vector_type(16))) __bf16 v16bf;
typedef __attribute__((ext_vector_type(8)))  float  v8f;

// ---------- helpers ----------
__device__ inline __bf16 f2bf(float f) {
  union { float f; unsigned u; } v; v.f = f;
  unsigned r = v.u + 0x7FFFu + ((v.u >> 16) & 1u);   // round-to-nearest-even
  unsigned short h = (unsigned short)(r >> 16);
  __bf16 b; __builtin_memcpy(&b, &h, 2);
  return b;
}

__device__ inline float sigmoidf_(float x) { return 1.0f / (1.0f + __expf(-x)); }

// CDNA5 async memory->LDS DMA (tracked by ASYNCcnt, not LOADcnt/DScnt).
// GV addressing: per-lane 64-bit global address, per-lane 32-bit LDS byte address.
__device__ __forceinline__ void async_b128(unsigned lds_off, const void* gaddr) {
  asm volatile("global_load_async_to_lds_b128 %0, %1, off"
               :: "v"(lds_off), "v"(gaddr)
               : "memory");
}
template <int N>
__device__ __forceinline__ void wait_asynccnt() {
  asm volatile("s_wait_asynccnt %0" :: "n"(N) : "memory");
}
// low 32 bits of a flat shared pointer == LDS byte offset
__device__ __forceinline__ unsigned lds_off32(const void* p) {
  return (unsigned)(uintptr_t)p;
}

// ---------- f32 -> bf16 conversion (weights, hidden) ----------
__global__ __launch_bounds__(256)
void cvt_f32_bf16(const float* __restrict__ s, __bf16* __restrict__ d, long n) {
  long i = (long)blockIdx.x * blockDim.x + threadIdx.x;
  long stride = (long)gridDim.x * blockDim.x;
  for (; i < n; i += stride) d[i] = f2bf(s[i]);
}

// ---------- embedding gather (all timesteps up front) ----------
__global__ __launch_bounds__(256)
void embed_kernel(const int* __restrict__ tok, const float* __restrict__ emb,
                  __bf16* __restrict__ Xb) {
  const int row = blockIdx.x;                 // t*BATCH + b
  const int t = tok[row];
  const float scale = 32.0f;                  // sqrt(EMB)
  for (int e = threadIdx.x; e < EMB; e += blockDim.x)
    Xb[(size_t)row * EMB + e] = f2bf(emb[(size_t)t * EMB + e] * scale);
}

// ---------- generic fused dual-GEMM:  out = A1@W1^T [+ A2@W2^T] [+ bias] [+ Cinit] ----------
// A: M x K (bf16, row major), W: N x K (bf16, row major) -> computes A @ W^T.
// BM=64, BN=128, BK=32; 256 threads = 8 waves; wave w owns a 16-wide N column,
// computing four 16x16 tiles via v_wmma_f32_16x16x32_bf16.
// K-tiles are double-buffered in LDS via global_load_async_to_lds_b128:
// each wave issues exactly 3 async b128 ops per stage, so s_wait_asynccnt 3
// after issuing stage k+1 guarantees stage k's data has landed.
#define BM 64
#define BN 128
#define BK 32
#define LDA 40   // padded bf16 row stride (80B rows -> 16B aligned, bank-spread)
#define LDB 40

__global__ __launch_bounds__(256)
void gemm_bf16_wmma(const __bf16* __restrict__ A1, const __bf16* __restrict__ W1,
                    const __bf16* __restrict__ A2, const __bf16* __restrict__ W2,
                    const float* __restrict__ bias, const float* __restrict__ Cinit,
                    float* __restrict__ outF, __bf16* __restrict__ outB,
                    int M, int N, int K, int ntOut) {
  __shared__ __align__(16) __bf16 As[2][BM * LDA];
  __shared__ __align__(16) __bf16 Ws[2][BN * LDB];

  const int tid  = threadIdx.x;
  const int wave = tid >> 5;
  const int lane = tid & 31;
  const int half = lane >> 4;
  const int l16  = lane & 15;
  const int m0   = blockIdx.y * BM;
  const int n0   = blockIdx.x * BN;

  // staging coordinates: A tile 64x32 -> 1 b128/thread; W tile 128x32 -> 2 b128/thread
  const int arow = tid >> 2,            acol = (tid & 3) * 8;
  const int w0row = tid >> 2,           w0col = (tid & 3) * 8;
  const int w1row = (tid + 256) >> 2,   w1col = (tid & 3) * 8;
  unsigned aOff[2], wOff0[2], wOff1[2];
#pragma unroll
  for (int b = 0; b < 2; ++b) {
    aOff[b]  = lds_off32(&As[b][arow * LDA + acol]);
    wOff0[b] = lds_off32(&Ws[b][w0row * LDB + w0col]);
    wOff1[b] = lds_off32(&Ws[b][w1row * LDB + w1col]);
  }

  const v8f zero = {0.f, 0.f, 0.f, 0.f, 0.f, 0.f, 0.f, 0.f};
  v8f c[4];
#pragma unroll
  for (int m = 0; m < 4; ++m) c[m] = zero;

  const int kIters = K / BK;
  const int total  = A2 ? 2 * kIters : kIters;

  // prologue: stage iteration 0 into buffer 0
  {
    const __bf16* A = A1;
    const __bf16* W = W1;
    async_b128(aOff[0],  A + (size_t)(m0 + arow) * K + acol);
    async_b128(wOff0[0], W + (size_t)(n0 + w0row) * K + w0col);
    async_b128(wOff1[0], W + (size_t)(n0 + w1row) * K + w1col);
  }

  int buf = 0;
  for (int it = 0; it < total; ++it) {
    // issue next stage into the other buffer while this one is consumed
    if (it + 1 < total) {
      const int nit = it + 1;
      const __bf16* A = (nit < kIters) ? A1 : A2;
      const __bf16* W = (nit < kIters) ? W1 : W2;
      const int kk = (nit < kIters ? nit : nit - kIters) * BK;
      const int nb = buf ^ 1;
      async_b128(aOff[nb],  A + (size_t)(m0 + arow) * K + kk + acol);
      async_b128(wOff0[nb], W + (size_t)(n0 + w0row) * K + kk + w0col);
      async_b128(wOff1[nb], W + (size_t)(n0 + w1row) * K + kk + w1col);
      wait_asynccnt<3>();     // stage `it` (issued earlier) has landed
    } else {
      wait_asynccnt<0>();
    }
    __syncthreads();          // all waves' deposits visible

    // B fragment (32x16, K x N), documented 16-bit layout:
    // lane: column = lane&15; v0..3 -> K = half*8 + 2v; v4..7 -> K = 16 + half*8 + 2(v-4)
    union { v16bf v; unsigned u[8]; } bfr;
    const int wn = wave * 16 + l16;
#pragma unroll
    for (int v = 0; v < 8; ++v)
      bfr.u[v] = *(const unsigned*)&Ws[buf][wn * LDB + (v >> 2) * 16 + half * 8 + 2 * (v & 3)];

#pragma unroll
    for (int m = 0; m < 4; ++m) {
      // A fragment (16x32), documented layout:
      // row = lane&15; v0..3 -> K = half*8 + 2v; v4..7 -> K = 16 + half*8 + 2(v-4)
      union { v16bf v; unsigned u[8]; } afr;
      const int row = m * 16 + l16;
#pragma unroll
      for (int v = 0; v < 4; ++v)
        afr.u[v] = *(const unsigned*)&As[buf][row * LDA + half * 8 + 2 * v];
#pragma unroll
      for (int v = 0; v < 4; ++v)
        afr.u[4 + v] = *(const unsigned*)&As[buf][row * LDA + 16 + half * 8 + 2 * v];

      c[m] = __builtin_amdgcn_wmma_f32_16x16x32_bf16(
          false, afr.v, false, bfr.v, (short)0, c[m], false, false);
    }
    __syncthreads();          // everyone done reading before this buffer is refilled
    buf ^= 1;
  }

  // epilogue: C/D layout -> row = m0 + mtile*16 + half*8 + r, col = n0 + wave*16 + (lane&15)
  const int coln = n0 + wave * 16 + l16;
#pragma unroll
  for (int m = 0; m < 4; ++m) {
#pragma unroll
    for (int r = 0; r < 8; ++r) {
      const int row = m0 + m * 16 + half * 8 + r;
      const size_t idx = (size_t)row * N + coln;
      float val = c[m][r];
      if (bias)  val += bias[coln];
      if (Cinit) val += Cinit[idx];
      if (outF) {
        if (ntOut) __builtin_nontemporal_store(val, &outF[idx]);  // don't sweep L2
        else       outF[idx] = val;
      }
      if (outB) outB[idx] = f2bf(val);
    }
  }
}

// ---------- elementwise: reset gate ----------
__global__ __launch_bounds__(256)
void ew_reset(const float* __restrict__ pre, const float* __restrict__ H,
              __bf16* __restrict__ preB, __bf16* __restrict__ rHb, int n) {
  int i = blockIdx.x * blockDim.x + threadIdx.x;
  if (i >= n) return;
  const float p = pre[i];
  const float r = sigmoidf_(p);
  preB[i] = f2bf(p);            // pre-activation feeds next layer
  rHb[i]  = f2bf(r * H[i]);     // r ⊙ h_prev for candidate GEMM
}

// ---------- elementwise: forget/candidate + state update ----------
__global__ __launch_bounds__(256)
void ew_update(const float* __restrict__ fpre, const float* __restrict__ hpre,
               float* __restrict__ H, __bf16* __restrict__ Hb,
               __bf16* __restrict__ fB, __bf16* __restrict__ hB,
               __bf16* __restrict__ hallB, int n) {
  int i = blockIdx.x * blockDim.x + threadIdx.x;
  if (i >= n) return;
  const float fp = fpre[i];
  const float hp = hpre[i];
  const float z = sigmoidf_(fp);
  const float h = (1.0f - z) * H[i] + z * tanhf(hp);
  H[i]  = h;
  Hb[i] = f2bf(h);
  fB[i] = f2bf(fp);
  hB[i] = f2bf(hp);
  if (hallB) hallB[i] = f2bf(h);
}

// ---------- copy final hidden state (f32) into the output tail ----------
__global__ __launch_bounds__(256)
void copy_f32(const float* __restrict__ s, float* __restrict__ d, int n) {
  int i = blockIdx.x * blockDim.x + threadIdx.x;
  if (i < n) d[i] = s[i];
}

// ---------- host orchestration ----------
extern "C" void kernel_launch(void* const* d_in, const int* in_sizes, int n_in,
                              void* d_out, int out_size, void* d_ws, size_t ws_size,
                              hipStream_t stream) {
  (void)in_sizes; (void)n_in; (void)out_size; (void)ws_size;

  const int*   tok    = (const int*)d_in[0];
  const float* hidden = (const float*)d_in[1];
  const float* emb    = (const float*)d_in[2];
  const float* W_in   = (const float*)d_in[3];
  const float* Wr1    = (const float*)d_in[4];
  const float* Wf1    = (const float*)d_in[5];
  const float* Wh1    = (const float*)d_in[6];
  const float* Ur     = (const float*)d_in[7];
  const float* br     = (const float*)d_in[8];
  const float* Uf     = (const float*)d_in[9];
  const float* bfb    = (const float*)d_in[10];
  const float* Uh     = (const float*)d_in[11];
  const float* bh     = (const float*)d_in[12];
  const float* outW   = (const float*)d_in[13];
  const float* outb   = (const float*)d_in[14];

  float* logits = (float*)d_out;                                   // (SEQ*BATCH, VOCAB)
  float* hfin   = logits + (size_t)SEQ * BATCH * VOCAB;            // (L*BATCH, HID)

  // ---- workspace carve-up ----
  char* wsb = (char*)d_ws;
  size_t off = 0;
  auto alloc = [&](size_t bytes) -> char* {
    off = (off + 255) & ~(size_t)255;
    char* p = wsb + off;
    off += bytes;
    return p;
  };
  const size_t MM = (size_t)HID * HID;          // 1M elements
  __bf16* wb_Win  = (__bf16*)alloc(MM * 2);
  __bf16* wb_Wr1  = (__bf16*)alloc(MM * 2);
  __bf16* wb_Wf1  = (__bf16*)alloc(MM * 2);
  __bf16* wb_Wh1  = (__bf16*)alloc(MM * 2);
  __bf16* wb_Ur   = (__bf16*)alloc(L * MM * 2);
  __bf16* wb_Uf   = (__bf16*)alloc(L * MM * 2);
  __bf16* wb_Uh   = (__bf16*)alloc(L * MM * 2);
  __bf16* wb_outW = (__bf16*)alloc((size_t)VOCAB * HID * 2);
  __bf16* Xb      = (__bf16*)alloc((size_t)SEQ * BATCH * EMB * 2);
  __bf16* Hallb   = (__bf16*)alloc((size_t)SEQ * BATCH * HID * 2);
  float*  Hf      = (float*)alloc((size_t)L * BATCH * HID * 4);
  __bf16* Hb      = (__bf16*)alloc((size_t)L * BATCH * HID * 2);
  float*  xw      = (float*)alloc((size_t)BATCH * HID * 4);
  float*  rpre    = (float*)alloc((size_t)BATCH * HID * 4);
  float*  fpre    = (float*)alloc((size_t)BATCH * HID * 4);
  float*  hpre    = (float*)alloc((size_t)BATCH * HID * 4);
  __bf16* rB      = (__bf16*)alloc((size_t)BATCH * HID * 2);
  __bf16* fB      = (__bf16*)alloc((size_t)BATCH * HID * 2);
  __bf16* hB      = (__bf16*)alloc((size_t)BATCH * HID * 2);
  __bf16* rHb     = (__bf16*)alloc((size_t)BATCH * HID * 2);

  const int n1 = BATCH * HID;
  const dim3 blk(256);
  const dim3 ewg((n1 + 255) / 256);

  auto cvt = [&](const float* s, __bf16* d, size_t n) {
    cvt_f32_bf16<<<dim3(1024), blk, 0, stream>>>(s, d, (long)n);
  };
  auto gemm = [&](const __bf16* A1, const __bf16* W1, const __bf16* A2, const __bf16* W2,
                  const float* bias, const float* Cinit, float* oF, __bf16* oB,
                  int M, int N, int K, int nt) {
    gemm_bf16_wmma<<<dim3(N / BN, M / BM), blk, 0, stream>>>(
        A1, W1, A2, W2, bias, Cinit, oF, oB, M, N, K, nt);
  };

  // ---- phase 0: weight conversion + embedding gather + hidden init ----
  cvt(W_in, wb_Win, MM);
  cvt(Wr1, wb_Wr1, MM);
  cvt(Wf1, wb_Wf1, MM);
  cvt(Wh1, wb_Wh1, MM);
  cvt(Ur, wb_Ur, L * MM);
  cvt(Uf, wb_Uf, L * MM);
  cvt(Uh, wb_Uh, L * MM);
  cvt(outW, wb_outW, (size_t)VOCAB * HID);
  embed_kernel<<<dim3(SEQ * BATCH), blk, 0, stream>>>(tok, emb, Xb);
  hipMemcpyAsync(Hf, hidden, (size_t)L * BATCH * HID * 4, hipMemcpyDeviceToDevice, stream);
  cvt(Hf, Hb, (size_t)L * BATCH * HID);

  float*  Hf0 = Hf;            float*  Hf1 = Hf + n1;
  __bf16* Hb0 = Hb;            __bf16* Hb1 = Hb + n1;
  __bf16* Ur0 = wb_Ur;         __bf16* Ur1 = wb_Ur + MM;
  __bf16* Uf0 = wb_Uf;         __bf16* Uf1 = wb_Uf + MM;
  __bf16* Uh0 = wb_Uh;         __bf16* Uh1 = wb_Uh + MM;

  // ---- phase 1: sequential GRU scan (bf16 weights stay resident in L2) ----
  for (int t = 0; t < SEQ; ++t) {
    const __bf16* Xt = Xb + (size_t)t * BATCH * EMB;

    // layer 0 — shared input projection computed once, reused by all 3 gates
    gemm(Xt,  wb_Win, nullptr, nullptr, nullptr, nullptr, xw,   nullptr, BATCH, HID, EMB, 0);
    gemm(Hb0, Ur0,    nullptr, nullptr, br,       xw,     rpre, nullptr, BATCH, HID, HID, 0);
    gemm(Hb0, Uf0,    nullptr, nullptr, bfb,      xw,     fpre, nullptr, BATCH, HID, HID, 0);
    ew_reset<<<ewg, blk, 0, stream>>>(rpre, Hf0, rB, rHb, n1);
    gemm(rHb, Uh0,    nullptr, nullptr, bh,       xw,     hpre, nullptr, BATCH, HID, HID, 0);
    ew_update<<<ewg, blk, 0, stream>>>(fpre, hpre, Hf0, Hb0, fB, hB, nullptr, n1);

    // layer 1 — pre-activations from layer 0 are the inputs (fused dual GEMMs)
    gemm(rB, wb_Wr1, Hb1, Ur1, br + HID,  nullptr, rpre, nullptr, BATCH, HID, HID, 0);
    gemm(fB, wb_Wf1, Hb1, Uf1, bfb + HID, nullptr, fpre, nullptr, BATCH, HID, HID, 0);
    ew_reset<<<ewg, blk, 0, stream>>>(rpre, Hf1, rB, rHb, n1);
    gemm(hB, wb_Wh1, rHb, Uh1, bh + HID,  nullptr, hpre, nullptr, BATCH, HID, HID, 0);
    ew_update<<<ewg, blk, 0, stream>>>(fpre, hpre, Hf1, Hb1, fB, hB,
                                       Hallb + (size_t)t * n1, n1);
  }

  // ---- phase 2: deferred vocab projection (75% of FLOPs), NT stores to HBM ----
  gemm(Hallb, wb_outW, nullptr, nullptr, outb, nullptr, logits, nullptr,
       SEQ * BATCH, VOCAB, HID, 1);

  // ---- final hidden state -> output tail ----
  copy_f32<<<dim3((L * n1 + 255) / 256), blk, 0, stream>>>(Hf, hfin, L * n1);
}
